// PatchBasedQuantumFilter_87084756894036
// MI455X (gfx1250) — matrix operand order repaired and checked by reading, not compiled, not run on hardware
//
#include <hip/hip_runtime.h>

typedef _Float16 v16h __attribute__((ext_vector_type(16)));
typedef float    v8f  __attribute__((ext_vector_type(8)));
typedef int      v4i  __attribute__((vector_size(16)));
typedef __attribute__((address_space(1))) v4i* v4i_gptr;   // global
typedef __attribute__((address_space(3))) v4i* v4i_lptr;   // LDS

#define N_QUBITS 8
#define N_LAYERS 4

#if defined(__has_builtin)
#if __has_builtin(__builtin_amdgcn_global_load_async_to_lds_b128)
#define HAVE_ASYNC_LDS 1
#endif
#if __has_builtin(__builtin_amdgcn_s_wait_asynccnt)
#define HAVE_WAIT_ASYNC 1
#endif
#endif

__device__ __forceinline__ void cswap(float& a, float& b) { float t = a; a = b; b = t; }

// ---------------------------------------------------------------------------
// K1: build the fixed 256x256 circuit unitary U, one wave per column.
// State: 8 complex amps per lane, amp index j = lane*8 + u.
//   qubits 0..2 -> bits of u (intra-lane), qubits 3..7 -> lane bits (shfl_xor).
// Output Bp: packed WMMA-B fragments (f16), layout [kb(8)][nb(32)][lane(32)][h(16)]:
//   lane <  16 : N = lane,      K = kb*32 + h
//   lane >= 16 : N = lane - 16, K = kb*32 + 16 + h
// Column n<256 = Re U[j=n, k], n>=256 = Im U[j=n-256, k].  Total 256 KB.
// ---------------------------------------------------------------------------
__global__ __launch_bounds__(128) void build_u_kernel(
    const float* __restrict__ w, _Float16* __restrict__ Bp) {
  const int lane = threadIdx.x & 31;
  const int k = blockIdx.x * 4 + (threadIdx.x >> 5);   // column index 0..255

  float re[8], im[8];
#pragma unroll
  for (int u = 0; u < 8; ++u) {
    re[u] = ((lane * 8 + u) == k) ? 1.0f : 0.0f;
    im[u] = 0.0f;
  }

#pragma unroll
  for (int layer = 0; layer < N_LAYERS; ++layer) {
    // ---- CNOT ladder: (0,1),(1,2),(2,3),(3,4),(4,5),(5,6),(6,7),(7,0)
    cswap(re[1], re[3]); cswap(im[1], im[3]);
    cswap(re[5], re[7]); cswap(im[5], im[7]);
    cswap(re[2], re[6]); cswap(im[2], im[6]);
    cswap(re[3], re[7]); cswap(im[3], im[7]);
#pragma unroll
    for (int u = 4; u < 8; ++u) {
      re[u] = __shfl_xor(re[u], 1, 32);
      im[u] = __shfl_xor(im[u], 1, 32);
    }
#pragma unroll
    for (int g = 0; g < 4; ++g) {
      const int cbit = 1 << g;
      const int tmask = 2 << g;
      const bool sel = (lane & cbit) != 0;
#pragma unroll
      for (int u = 0; u < 8; ++u) {
        float pr = __shfl_xor(re[u], tmask, 32);
        float pi = __shfl_xor(im[u], tmask, 32);
        re[u] = sel ? pr : re[u];
        im[u] = sel ? pi : im[u];
      }
    }
    {
      const bool sel = (lane & 16) != 0;
#pragma unroll
      for (int u = 0; u < 8; u += 2) {
        float r0 = re[u], r1 = re[u + 1];
        float i0 = im[u], i1 = im[u + 1];
        re[u] = sel ? r1 : r0; re[u + 1] = sel ? r0 : r1;
        im[u] = sel ? i1 : i0; im[u + 1] = sel ? i0 : i1;
      }
    }
    // ---- per-wire RZ then RY
#pragma unroll
    for (int q = 0; q < N_QUBITS; ++q) {
      const float tz = w[(layer * 8 + q) * 2 + 1];
      const float ty = w[(layer * 8 + q) * 2 + 0];
      float sz, cz; __sincosf(0.5f * tz, &sz, &cz);
      float sy, cy; __sincosf(0.5f * ty, &sy, &cy);
      if (q < 3) {
#pragma unroll
        for (int u = 0; u < 8; ++u) {
          float sg = ((u >> q) & 1) ? sz : -sz;
          float nr = cz * re[u] - sg * im[u];
          float ni = cz * im[u] + sg * re[u];
          re[u] = nr; im[u] = ni;
        }
      } else {
        const float sg = ((lane >> (q - 3)) & 1) ? sz : -sz;
#pragma unroll
        for (int u = 0; u < 8; ++u) {
          float nr = cz * re[u] - sg * im[u];
          float ni = cz * im[u] + sg * re[u];
          re[u] = nr; im[u] = ni;
        }
      }
      if (q < 3) {
        const int m = 1 << q;
#pragma unroll
        for (int u = 0; u < 8; ++u) {
          if ((u & m) == 0) {
            const int v = u | m;
            float r0 = cy * re[u] - sy * re[v];
            float r1 = sy * re[u] + cy * re[v];
            float i0 = cy * im[u] - sy * im[v];
            float i1 = sy * im[u] + cy * im[v];
            re[u] = r0; re[v] = r1; im[u] = i0; im[v] = i1;
          }
        }
      } else {
        const int lm = 1 << (q - 3);
        const float sg = (lane & lm) ? sy : -sy;
#pragma unroll
        for (int u = 0; u < 8; ++u) {
          float pr = __shfl_xor(re[u], lm, 32);
          float pi = __shfl_xor(im[u], lm, 32);
          re[u] = cy * re[u] + sg * pr;
          im[u] = cy * im[u] + sg * pi;
        }
      }
    }
  }

  // ---- scatter this column into packed B fragments (16 x 2-byte stores/lane)
  const int kb = k >> 5;
  const int kl = k & 31;
  const int hgrp = (kl >> 4) & 1;
  const int h = kl & 15;
#pragma unroll
  for (int u = 0; u < 8; ++u) {
    const int n_re = lane * 8 + u;
    const int nb = n_re >> 4;
    const int lf = (n_re & 15) | (hgrp << 4);
    Bp[(((kb * 32 + nb) * 32) + lf) * 16 + h] = (_Float16)re[u];
    const int nbi = nb + 16;
    Bp[(((kb * 32 + nbi) * 32) + lf) * 16 + h] = (_Float16)im[u];
  }
}

// ---------------------------------------------------------------------------
// K2: encode + GEMM + fused <Z_i> reduction.
// 64 blocks x 256 threads (8 waves, one 16-sample M-tile each).
// The whole packed B (256 KB) is staged in LDS once per block via the
// gfx1250 async global->LDS path (ASYNCcnt), then each wave runs 32 N-blocks
// x 8 K-blocks of v_wmma_f32_16x16x32_f16 reading B fragments via ds_load.
// ---------------------------------------------------------------------------
__global__ __launch_bounds__(256) void qforward_kernel(
    const float* __restrict__ x, const float* __restrict__ w,
    const _Float16* __restrict__ Bp, float* __restrict__ out) {
  extern __shared__ _Float16 Bs[];   // 131072 halves = 256 KB (dynamic LDS)

  const int tid = threadIdx.x;
  const int lane = tid & 31;
  const int mb = blockIdx.x * 8 + (tid >> 5);  // 0..511
  const int b = mb * 16 + (lane & 15);         // sample row (M = lane%16)

  // ---- stage B into LDS: 16384 x 16B chunks, 64 per thread, coalesced
#pragma unroll 4
  for (int t = tid; t < 16384; t += 256) {
#if defined(HAVE_ASYNC_LDS)
    __builtin_amdgcn_global_load_async_to_lds_b128(
        (v4i_gptr)(const_cast<_Float16*>(Bp) + t * 8),
        (v4i_lptr)(Bs + t * 8),
        /*offset=*/0, /*cpol=*/0);
#else
    *reinterpret_cast<int4*>(Bs + t * 8) =
        *reinterpret_cast<const int4*>(Bp + t * 8);
#endif
  }
#if defined(HAVE_ASYNC_LDS)
#if defined(HAVE_WAIT_ASYNC)
  __builtin_amdgcn_s_wait_asynccnt(0);
#else
  asm volatile("s_wait_asynccnt 0x0" ::: "memory");
#endif
#endif

  // ---- encoding angles + A fragments (after async fill has completed)
  float cw[8], sw[8];
#pragma unroll
  for (int i = 0; i < 8; ++i) {
    float ph = 0.5f * (x[b * 8 + i] + w[2 * i]);
    __sincosf(ph, &sw[i], &cw[i]);
  }

  v16h afr[8];
#pragma unroll
  for (int kb = 0; kb < 8; ++kb) {
#pragma unroll
    for (int h = 0; h < 16; ++h) {
      const int kl = h | (lane & 16);
      const int k = kb * 32 + kl;
      float v = (k & 1) ? sw[0] : cw[0];
#pragma unroll
      for (int i = 1; i < 8; ++i) v *= ((k >> i) & 1) ? sw[i] : cw[i];
      afr[kb][h] = (_Float16)v;
    }
  }

  __syncthreads();   // B fully resident in LDS for all 8 waves

  float acc[8][8];
#pragma unroll
  for (int r = 0; r < 8; ++r)
#pragma unroll
    for (int i = 0; i < 8; ++i) acc[r][i] = 0.0f;

  for (int nb = 0; nb < 32; ++nb) {
    v8f c = {0.f, 0.f, 0.f, 0.f, 0.f, 0.f, 0.f, 0.f};
#pragma unroll
    for (int kb = 0; kb < 8; ++kb) {
      const v16h bfr = *reinterpret_cast<const v16h*>(
          Bs + (((kb * 32 + nb) * 32) + lane) * 16);   // ds_load_b128 x2
      c = __builtin_amdgcn_wmma_f32_16x16x32_f16(
          false, afr[kb], false, bfr, (short)0, c, false, false);
    }
    // fused readout: acc[r][i] += sign_i(j) * amp^2
    const int n = nb * 16 + (lane & 15);
    const int j = n & 255;
#pragma unroll
    for (int r = 0; r < 8; ++r) {
      const float p = c[r] * c[r];
#pragma unroll
      for (int i = 0; i < 8; ++i)
        acc[r][i] += ((j >> i) & 1) ? -p : p;
    }
  }

  // reduce over the 16 lanes of each half (rows M=r for lanes<16, M=r+8 for >=16)
#pragma unroll
  for (int mask = 1; mask <= 8; mask <<= 1)
#pragma unroll
    for (int r = 0; r < 8; ++r)
#pragma unroll
      for (int i = 0; i < 8; ++i)
        acc[r][i] += __shfl_xor(acc[r][i], mask, 32);

  if ((lane & 15) == 0) {
    const int rowbase = mb * 16 + ((lane >> 4) << 3);
#pragma unroll
    for (int r = 0; r < 8; ++r)
#pragma unroll
      for (int i = 0; i < 8; ++i)
        out[(rowbase + r) * 8 + i] = acc[r][i];
  }
}

extern "C" void kernel_launch(void* const* d_in, const int* in_sizes, int n_in,
                              void* d_out, int out_size, void* d_ws, size_t ws_size,
                              hipStream_t stream) {
  (void)in_sizes; (void)n_in; (void)out_size; (void)ws_size;
  const float* x = (const float*)d_in[0];   // (8192, 8) f32
  const float* w = (const float*)d_in[1];   // (4, 8, 2) f32
  _Float16* Bp = (_Float16*)d_ws;           // packed U: 256 KB
  float* out = (float*)d_out;               // (8192, 8) f32

  build_u_kernel<<<64, 128, 0, stream>>>(w, Bp);               // 256 waves
  qforward_kernel<<<64, 256, 262144, stream>>>(x, w, Bp, out); // 512 waves, B in LDS
}
